// QuantMlp_86835648790560
// MI455X (gfx1250) — 8-bit matrix operand order repaired and checked by reading, compiled, not run on hardware
//
#include <hip/hip_runtime.h>
#include <math.h>


typedef __attribute__((ext_vector_type(8))) int v8i;

#define TB 256

#ifndef __has_builtin
#define __has_builtin(x) 0
#endif

#if __has_builtin(__builtin_amdgcn_global_load_async_to_lds_b128)
#define HAVE_ASYNC_COPY 1
typedef __attribute__((__vector_size__(4 * sizeof(int)))) int async_v4i;
__device__ __forceinline__ void async_copy16(const void* g, void* l) {
  __builtin_amdgcn_global_load_async_to_lds_b128((async_v4i*)g, (async_v4i*)l, 0, 0);
}
__device__ __forceinline__ void async_wait0() {
#if __has_builtin(__builtin_amdgcn_s_wait_asynccnt)
  __builtin_amdgcn_s_wait_asynccnt(0);
#else
  asm volatile("s_wait_asynccnt 0x0" ::: "memory");
#endif
}
#else
#define HAVE_ASYNC_COPY 0
#endif

__global__ void zero4_kernel(unsigned* p) {
  if (threadIdx.x < 4) p[threadIdx.x] = 0u;
}

__global__ void absmax_kernel(const float* __restrict__ v, size_t n, unsigned* slot) {
  __shared__ float red[TB];
  float m = 0.f;
  for (size_t i = (size_t)blockIdx.x * TB + threadIdx.x; i < n; i += (size_t)gridDim.x * TB)
    m = fmaxf(m, fabsf(v[i]));
  red[threadIdx.x] = m;
  __syncthreads();
  for (int s = TB / 2; s > 0; s >>= 1) {
    if (threadIdx.x < s) red[threadIdx.x] = fmaxf(red[threadIdx.x], red[threadIdx.x + s]);
    __syncthreads();
  }
  if (threadIdx.x == 0) atomicMax(slot, __float_as_uint(red[0]));
}

__global__ void ln_absmax_kernel(const float* __restrict__ x, const float* __restrict__ gamma,
                                 const float* __restrict__ beta, float* __restrict__ xn,
                                 unsigned* slot, int D) {
  __shared__ float red[TB];
  const size_t row = blockIdx.x;
  const float* xr = x + row * (size_t)D;
  float* o = xn + row * (size_t)D;
  const int t = threadIdx.x;

  float sum = 0.f, ss = 0.f;
  for (int i = t; i < D; i += TB) { float v = xr[i]; sum += v; ss += v * v; }

  red[t] = sum; __syncthreads();
  for (int s = TB / 2; s > 0; s >>= 1) { if (t < s) red[t] += red[t + s]; __syncthreads(); }
  const float mean = red[0] / (float)D;
  __syncthreads();

  red[t] = ss; __syncthreads();
  for (int s = TB / 2; s > 0; s >>= 1) { if (t < s) red[t] += red[t + s]; __syncthreads(); }
  const float var = fmaxf(red[0] / (float)D - mean * mean, 0.f);
  __syncthreads();

  const float rstd = rsqrtf(var + 1e-5f);
  float m = 0.f;
  for (int i = t; i < D; i += TB) {
    float v = (xr[i] - mean) * rstd * gamma[i] + beta[i];
    o[i] = v;
    m = fmaxf(m, fabsf(v));
  }
  red[t] = m; __syncthreads();
  for (int s = TB / 2; s > 0; s >>= 1) { if (t < s) red[t] = fmaxf(red[t], red[t + s]); __syncthreads(); }
  if (t == 0) atomicMax(slot, __float_as_uint(red[0]));
}

__global__ void quant_kernel(const float* __restrict__ v, signed char* __restrict__ q,
                             const unsigned* slot, float denom, float qmin, float qmax, size_t n4) {
  const float s = fmaxf(__uint_as_float(*slot) / denom, 1e-12f);
  const float inv = 1.0f / s;
  for (size_t i = (size_t)blockIdx.x * TB + threadIdx.x; i < n4; i += (size_t)gridDim.x * TB) {
    float4 f = ((const float4*)v)[i];
    char4 c;
    c.x = (signed char)fminf(fmaxf(rintf(f.x * inv), qmin), qmax);
    c.y = (signed char)fminf(fmaxf(rintf(f.y * inv), qmin), qmax);
    c.z = (signed char)fminf(fmaxf(rintf(f.z * inv), qmin), qmax);
    c.w = (signed char)fminf(fmaxf(rintf(f.w * inv), qmin), qmax);
    ((char4*)q)[i] = c;
  }
}

__global__ __launch_bounds__(TB)
void gemm_i8_kernel(const signed char* __restrict__ A, const signed char* __restrict__ B,
                    const float* __restrict__ bias,
                    const unsigned* sa_slot, const unsigned* sw_slot,
                    float* __restrict__ C, unsigned* hmax_slot,
                    int N, int K, int do_gelu) {
  __shared__ __align__(16) signed char As[2][128 * 64];
  __shared__ __align__(16) signed char Bs[2][128 * 64];
  __shared__ float red[TB];

  const int t     = threadIdx.x;
  const int lane  = t & 31;
  const int wave  = t >> 5;
  const int waveM = wave >> 2;
  const int waveN = wave & 3;
  const int mrow  = lane & 15;
  const int hh    = lane >> 4;

  const size_t blockM = (size_t)blockIdx.y * 128;
  const int    blockN = blockIdx.x * 128;

  int srow[2], schk[2];
#pragma unroll
  for (int i = 0; i < 2; ++i) { int idx = t + i * TB; srow[i] = idx >> 2; schk[i] = (idx & 3) << 4; }

  v8i acc[4][2] = {};

  auto compute_tile = [&](const signed char* Asb, const signed char* Bsb) {
    v8i afrag[4], bfrag[2];
#pragma unroll
    for (int mt = 0; mt < 4; ++mt) {
      const signed char* base = Asb + (waveM * 64 + mt * 16 + mrow) * 64 + hh * 16;
#pragma unroll
      for (int j = 0; j < 2; ++j) {
        int4 d = *(const int4*)(base + 32 * j);
        afrag[mt][4 * j]     = d.x;
        afrag[mt][4 * j + 1] = d.y;
        afrag[mt][4 * j + 2] = d.z;
        afrag[mt][4 * j + 3] = d.w;
      }
    }
#pragma unroll
    for (int nt = 0; nt < 2; ++nt) {
      const signed char* base = Bsb + (waveN * 32 + nt * 16 + mrow) * 64 + hh * 16;
      int4 d0 = *(const int4*)(base);
      int4 d1 = *(const int4*)(base + 32);
      bfrag[nt][0] = d0.x; bfrag[nt][1] = d0.y; bfrag[nt][2] = d0.z; bfrag[nt][3] = d0.w;
      bfrag[nt][4] = d1.x; bfrag[nt][5] = d1.y; bfrag[nt][6] = d1.z; bfrag[nt][7] = d1.w;
    }
#pragma unroll
    for (int mt = 0; mt < 4; ++mt)
#pragma unroll
      for (int nt = 0; nt < 2; ++nt)
        acc[mt][nt] = __builtin_amdgcn_wmma_i32_16x16x64_iu8(
             true, afrag[mt],  true, bfrag[nt],
            acc[mt][nt],  false,  false);
  };

#if HAVE_ASYNC_COPY
  auto issue_tile = [&](int k0, int buf) {
#pragma unroll
    for (int i = 0; i < 2; ++i) {
      async_copy16(A + (blockM + (size_t)srow[i]) * (size_t)K + k0 + schk[i],
                   &As[buf][srow[i] * 64 + schk[i]]);
      async_copy16(B + ((size_t)(blockN + srow[i])) * (size_t)K + k0 + schk[i],
                   &Bs[buf][srow[i] * 64 + schk[i]]);
    }
  };
  issue_tile(0, 0);
  async_wait0();
  __syncthreads();
  for (int k0 = 0; k0 < K; k0 += 64) {
    const int  buf  = (k0 >> 6) & 1;
    const bool more = (k0 + 64) < K;
    if (more) issue_tile(k0 + 64, buf ^ 1);
    compute_tile(As[buf], Bs[buf]);
    if (more) async_wait0();
    __syncthreads();
  }
#else
  {
    int4 ra[2], rb[2];
#pragma unroll
    for (int i = 0; i < 2; ++i) {
      ra[i] = *(const int4*)(A + (blockM + (size_t)srow[i]) * (size_t)K + schk[i]);
      rb[i] = *(const int4*)(B + ((size_t)(blockN + srow[i])) * (size_t)K + schk[i]);
    }
#pragma unroll
    for (int i = 0; i < 2; ++i) {
      *(int4*)(&As[0][srow[i] * 64 + schk[i]]) = ra[i];
      *(int4*)(&Bs[0][srow[i] * 64 + schk[i]]) = rb[i];
    }
  }
  __syncthreads();
  for (int k0 = 0; k0 < K; k0 += 64) {
    const int  buf  = (k0 >> 6) & 1;
    const bool more = (k0 + 64) < K;
    int4 ra[2], rb[2];
    if (more) {
#pragma unroll
      for (int i = 0; i < 2; ++i) {
        ra[i] = *(const int4*)(A + (blockM + (size_t)srow[i]) * (size_t)K + (k0 + 64) + schk[i]);
        rb[i] = *(const int4*)(B + ((size_t)(blockN + srow[i])) * (size_t)K + (k0 + 64) + schk[i]);
      }
    }
    compute_tile(As[buf], Bs[buf]);
    if (more) {
#pragma unroll
      for (int i = 0; i < 2; ++i) {
        *(int4*)(&As[buf ^ 1][srow[i] * 64 + schk[i]]) = ra[i];
        *(int4*)(&Bs[buf ^ 1][srow[i] * 64 + schk[i]]) = rb[i];
      }
    }
    __syncthreads();
  }
#endif

  const float sA = fmaxf(__uint_as_float(*sa_slot) * (1.0f / 128.0f), 1e-12f);
  const float sW = fmaxf(__uint_as_float(*sw_slot) * (1.0f / 127.0f), 1e-12f);
  const float scale = sA * sW;

  float lmax = 0.f;
#pragma unroll
  for (int mt = 0; mt < 4; ++mt) {
#pragma unroll
    for (int nt = 0; nt < 2; ++nt) {
      const size_t m0 = blockM + waveM * 64 + mt * 16 + hh * 8;
      const int n = blockN + waveN * 32 + nt * 16 + mrow;
      const float bn = bias[n];
#pragma unroll
      for (int r = 0; r < 8; ++r) {
        float f = (float)acc[mt][nt][r] * scale + bn;
        if (do_gelu) f = 0.5f * f * (1.0f + erff(f * 0.70710678118654752f));
        C[(m0 + (size_t)r) * (size_t)N + n] = f;
        lmax = fmaxf(lmax, fabsf(f));
      }
    }
  }

  if (hmax_slot) {
    red[t] = lmax;
    __syncthreads();
    for (int s = TB / 2; s > 0; s >>= 1) {
      if (t < s) red[t] = fmaxf(red[t], red[t + s]);
      __syncthreads();
    }
    if (t == 0) atomicMax(hmax_slot, __float_as_uint(red[0]));
  }
}

extern "C" void kernel_launch(void* const* d_in, const int* in_sizes, int n_in,
                              void* d_out, int out_size, void* d_ws, size_t ws_size,
                              hipStream_t stream) {
  const float* x     = (const float*)d_in[0];
  const float* gamma = (const float*)d_in[1];
  const float* beta  = (const float*)d_in[2];
  const float* W1    = (const float*)d_in[3];
  const float* b1    = (const float*)d_in[4];
  const float* W2    = (const float*)d_in[5];
  const float* b2    = (const float*)d_in[6];

  const int    D = in_sizes[1];
  const int    H = in_sizes[4];
  const size_t M = (size_t)in_sizes[0] / D;

  char* ws = (char*)d_ws;
  unsigned* scal = (unsigned*)ws;
  size_t off = 256;
  float*       xn   = (float*)(ws + off);       off += M * (size_t)D * 4;
  float*       hbuf = (float*)(ws + off);       off += M * (size_t)H * 4;
  signed char* xq   = (signed char*)(ws + off); off += M * (size_t)D;
  signed char* hq   = (signed char*)(ws + off); off += M * (size_t)H;
  signed char* w1q  = (signed char*)(ws + off); off += (size_t)H * D;
  signed char* w2q  = (signed char*)(ws + off); off += (size_t)D * H;
  (void)off; (void)ws_size; (void)n_in; (void)out_size;

  zero4_kernel<<<1, 32, 0, stream>>>(scal);

  absmax_kernel<<<1024, TB, 0, stream>>>(W1, (size_t)H * D, scal + 1);
  absmax_kernel<<<1024, TB, 0, stream>>>(W2, (size_t)D * H, scal + 2);
  ln_absmax_kernel<<<(unsigned)M, TB, 0, stream>>>(x, gamma, beta, xn, scal + 0, D);

  quant_kernel<<<2048,  TB, 0, stream>>>(W1, w1q, scal + 1, 127.f, -127.f, 127.f, (size_t)H * D / 4);
  quant_kernel<<<2048,  TB, 0, stream>>>(W2, w2q, scal + 2, 127.f, -127.f, 127.f, (size_t)D * H / 4);
  quant_kernel<<<8192,  TB, 0, stream>>>(xn, xq,  scal + 0, 128.f, -128.f, 127.f, M * (size_t)D / 4);

  dim3 g1(H / 128, (unsigned)(M / 128));
  gemm_i8_kernel<<<g1, TB, 0, stream>>>(xq, w1q, b1, scal + 0, scal + 1,
                                        hbuf, scal + 3, H, D,  1);

  quant_kernel<<<16384, TB, 0, stream>>>(hbuf, hq, scal + 3, 128.f, -128.f, 127.f, M * (size_t)H / 4);
  dim3 g2(D / 128, (unsigned)(M / 128));
  gemm_i8_kernel<<<g2, TB, 0, stream>>>(hq, w2q, b2, scal + 3, scal + 2,
                                        (float*)d_out, nullptr, D, H,  0);
}
